// GCN_vanilla_4_layers_31593779430028
// MI455X (gfx1250) — matrix operand order repaired and checked by reading, not compiled, not run on hardware
//
#include <hip/hip_runtime.h>
#include <hip/hip_bf16.h>

// ---------------------------------------------------------------------------
// 4-layer GCN for MI455X (gfx1250, wave32, WMMA).
//   per layer:  Wt = transpose(W) as bf16          (tiny, once per layer)
//               P  = h @ W          (v_wmma_f32_16x16x32_bf16, LDS-staged B,
//                                    double-buffered, software-pipelined)
//               Q  = scatter-add of w_e * P[src_e]  (f32 atomics)
//               h' = bf16(relu(Q + b))              (last layer: f32, no relu)
// ---------------------------------------------------------------------------

typedef __attribute__((ext_vector_type(16))) __bf16 v16bf;
typedef __attribute__((ext_vector_type(8)))  __bf16 v8bf;
typedef __attribute__((ext_vector_type(8)))  float  v8f;

#define NNODES 50000
#define LSTRIDE 40   // padded bf16 stride per column: 80B -> conflict-free ds

__device__ __forceinline__ v16bf cat8(v8bf lo, v8bf hi) {
  v16bf r;
#pragma unroll
  for (int i = 0; i < 8; ++i) { r[i] = lo[i]; r[i + 8] = hi[i]; }
  return r;
}

// C[M,N] = A[M,K] @ W[K,N], A bf16 row-major, Wt = W^T bf16 [N,K].
// Block = 8 waves = 128 rows x 64 cols. B panel (64 cols x 32 K) staged in
// double-buffered LDS, shared by all waves. Each wave: 16x64 strip, 4
// accumulators, one A fragment reused across 4 WMMAs per K-step.
// Requires: M % 16 == 0, K % 32 == 0, N % 64 == 0.
__global__ __launch_bounds__(256) void gemm_wmma_bf16(
    const __bf16* __restrict__ A, const __bf16* __restrict__ Wt,
    float* __restrict__ C, int M, int K, int N)
{
  __shared__ __bf16 sB[2][64 * LSTRIDE];   // 2 x 5 KB

  const int tid  = threadIdx.x;
  const int lane = tid & 31;
  const int wave = tid >> 5;
  const int m0   = blockIdx.y * 128 + wave * 16;
  const int n0   = blockIdx.x * 64;

  const int half8  = (lane < 16) ? 0 : 8;    // A K-offset for this lane half
  const int half16 = (lane < 16) ? 0 : 16;   // unused; B's K-offset is half8 as well
  const int lrow   = lane & 15;

  // Clamp out-of-range waves' loads (only possible in the last y-block);
  // they still participate in fills/barriers, stores are skipped below.
  const int rowA = min(m0 + lrow, M - 1);
  const __bf16* __restrict__ Ar = A + (size_t)rowA * (size_t)K;

  // Cooperative B-panel fill: thread t moves one 16B chunk.
  const int fcol   = tid >> 2;               // 0..63 (column within panel)
  const int fchunk = (tid & 3) * 8;          // K sub-offset: 0,8,16,24
  const __bf16* __restrict__ gB =
      Wt + (size_t)(n0 + fcol) * (size_t)K + fchunk;
  __bf16* const lB = &sB[0][0] + fcol * LSTRIDE + fchunk;

  auto fill = [&](int buf, int k0) {
    *reinterpret_cast<v8bf*>(lB + buf * (64 * LSTRIDE)) =
        *reinterpret_cast<const v8bf*>(gB + k0);
  };
  auto loadA = [&](int k0) {
    const int ka = k0 + half8;
    return cat8(*reinterpret_cast<const v8bf*>(Ar + ka),
                *reinterpret_cast<const v8bf*>(Ar + ka + 16));
  };

  v8f acc0 = {}, acc1 = {}, acc2 = {}, acc3 = {};

  // Prologue: stage first panel + first A fragment.
  fill(0, 0);
  v16bf a_cur = loadA(0);
  __syncthreads();

  const int nIter = K >> 5;
  for (int i = 0; i < nIter; ++i) {
    const int cur = i & 1;

    v16bf a_next;
    if (i + 1 < nIter) {               // overlap next-step traffic with WMMA
      fill(cur ^ 1, (i + 1) << 5);
      a_next = loadA((i + 1) << 5);
      if (i + 2 < nIter) __builtin_prefetch(Ar + ((i + 2) << 5), 0, 0);
    }

    // B fragments from LDS: lane holds one column, K = half8+0..7 and half8+16..23.
    const __bf16* p0 = &sB[cur][(lrow)      * LSTRIDE + half8];
    const __bf16* p1 = &sB[cur][(16 + lrow) * LSTRIDE + half8];
    const __bf16* p2 = &sB[cur][(32 + lrow) * LSTRIDE + half8];
    const __bf16* p3 = &sB[cur][(48 + lrow) * LSTRIDE + half8];
    const v16bf b0 = cat8(*(const v8bf*)p0, *(const v8bf*)(p0 + 16));
    const v16bf b1 = cat8(*(const v8bf*)p1, *(const v8bf*)(p1 + 16));
    const v16bf b2 = cat8(*(const v8bf*)p2, *(const v8bf*)(p2 + 16));
    const v16bf b3 = cat8(*(const v8bf*)p3, *(const v8bf*)(p3 + 16));

    acc0 = __builtin_amdgcn_wmma_f32_16x16x32_bf16(false, a_cur, false, b0,
                                                   (short)0, acc0, false, false);
    acc1 = __builtin_amdgcn_wmma_f32_16x16x32_bf16(false, a_cur, false, b1,
                                                   (short)0, acc1, false, false);
    acc2 = __builtin_amdgcn_wmma_f32_16x16x32_bf16(false, a_cur, false, b2,
                                                   (short)0, acc2, false, false);
    acc3 = __builtin_amdgcn_wmma_f32_16x16x32_bf16(false, a_cur, false, b3,
                                                   (short)0, acc3, false, false);

    a_cur = a_next;
    __syncthreads();   // reads of sB[cur] done; fill of sB[cur^1] visible
  }

  if (m0 >= M) return;   // uniform per wave; M % 16 == 0

  // Store: accumulator VGPR r -> row m0 + r + half8, cols n0+{0,16,32,48}+lrow
#pragma unroll
  for (int r = 0; r < 8; ++r) {
    float* Crow = C + (size_t)(m0 + r + half8) * (size_t)N + n0 + lrow;
    Crow[0]  = acc0[r];
    Crow[16] = acc1[r];
    Crow[32] = acc2[r];
    Crow[48] = acc3[r];
  }
}

// Wt[n*K + k] = bf16(W[k*N + n])  (coalesced writes; W is tiny, L2-resident)
__global__ __launch_bounds__(256) void cvt_transpose_w(
    const float* __restrict__ W, __bf16* __restrict__ Wt, int K, int N)
{
  const unsigned i = blockIdx.x * 256u + threadIdx.x;
  if (i >= (unsigned)(K * N)) return;
  const int n = i / K;
  const int k = i % K;
  Wt[i] = (__bf16)W[(size_t)k * (size_t)N + n];
}

// dst[i] = bf16(src[i])
__global__ __launch_bounds__(256) void cvt_bf16(
    const float* __restrict__ src, __bf16* __restrict__ dst, unsigned total)
{
  const unsigned i = blockIdx.x * 256u + threadIdx.x;
  if (i < total) dst[i] = (__bf16)src[i];
}

// out[dst_e, :] += w_e * Xw[src_e, :]  — one thread per (edge, 4-feature
// chunk); consecutive threads share an edge -> coalesced gather + atomics.
__global__ __launch_bounds__(256) void spmm_scatter(
    const float* __restrict__ Xw, const int* __restrict__ src,
    const int* __restrict__ dst, const float* __restrict__ ew,
    float* __restrict__ out, unsigned E, int D, int lq /* log2(D/4) */)
{
  const unsigned idx   = blockIdx.x * 256u + threadIdx.x;
  const unsigned total = E << lq;
  if (idx >= total) return;

  const unsigned e = idx >> lq;
  const unsigned f = (idx & ((1u << lq) - 1u)) << 2;

  const int   s = src[e];
  const int   d = dst[e];
  const float w = ew[e];

  const float4 v =
      *reinterpret_cast<const float4*>(Xw + (size_t)s * (size_t)D + f);
  float* o = out + (size_t)d * (size_t)D + f;
  atomicAdd(o + 0, v.x * w);
  atomicAdd(o + 1, v.y * w);
  atomicAdd(o + 2, v.z * w);
  atomicAdd(o + 3, v.w * w);
}

// hb[i] = bf16(relu(agg[i] + bias[i % D]))   (hidden layers)
__global__ __launch_bounds__(256) void bias_relu_bf16(
    const float* __restrict__ agg, const float* __restrict__ bias,
    __bf16* __restrict__ hb, unsigned total, int D)
{
  const unsigned i = blockIdx.x * 256u + threadIdx.x;
  if (i >= total) return;
  const float v = agg[i] + bias[i & (unsigned)(D - 1)];
  hb[i] = (__bf16)fmaxf(v, 0.f);
}

// h[i] += bias[i % D]   in place, f32, no activation (final layer)
__global__ __launch_bounds__(256) void bias_f32(
    float* __restrict__ h, const float* __restrict__ bias,
    unsigned total, int D)
{
  const unsigned i = blockIdx.x * 256u + threadIdx.x;
  if (i >= total) return;
  h[i] = h[i] + bias[i & (unsigned)(D - 1)];
}

extern "C" void kernel_launch(void* const* d_in, const int* in_sizes, int n_in,
                              void* d_out, int out_size, void* d_ws, size_t ws_size,
                              hipStream_t stream)
{
  const float* x    = (const float*)d_in[0];
  const int*   esrc = (const int*)  d_in[1];
  const int*   edst = (const int*)  d_in[2];
  const float* ew   = (const float*)d_in[3];
  const float* Wl[4] = {(const float*)d_in[4], (const float*)d_in[6],
                        (const float*)d_in[8], (const float*)d_in[10]};
  const float* bl[4] = {(const float*)d_in[5], (const float*)d_in[7],
                        (const float*)d_in[9], (const float*)d_in[11]};
  const int Kl[4] = {512, 512, 256, 128};
  const int Nl[4] = {512, 256, 128, 64};

  const unsigned E = (unsigned)in_sizes[1];   // 800000

  // Workspace layout (16B-aligned offsets):
  char* w  = (char*)d_ws;
  float*  P  = (float*)(w);                   // [50000,512] f32   (X @ W)
  float*  Q  = (float*)(w + 102400000);       // [50000,512] f32   (aggregate)
  __bf16* Hb = (__bf16*)(w + 204800000);      // [50000,512] bf16  (activations)
  __bf16* Wt = (__bf16*)(w + 256000000);      // [512,512]   bf16  (W^T)

  // Layer-1 input: bf16 copy of x.
  {
    unsigned tot = (unsigned)NNODES * 512u;
    cvt_bf16<<<(tot + 255) / 256, 256, 0, stream>>>(x, Hb, tot);
  }

  for (int l = 0; l < 4; ++l) {
    const int K = Kl[l], N = Nl[l];
    const bool last = (l == 3);
    float* agg = last ? (float*)d_out : Q;

    // W^T in bf16.
    {
      unsigned tot = (unsigned)(K * N);
      cvt_transpose_w<<<(tot + 255) / 256, 256, 0, stream>>>(Wl[l], Wt, K, N);
    }

    // Dense GEMM: P = Hb @ W.
    {
      dim3 grid(N / 64, (NNODES + 127) / 128);
      gemm_wmma_bf16<<<grid, 256, 0, stream>>>(Hb, Wt, P, NNODES, K, N);
    }

    // Sparse aggregation with f32 atomics.
    hipMemsetAsync(agg, 0, (size_t)NNODES * (size_t)N * sizeof(float), stream);
    {
      const int lq = __builtin_ctz((unsigned)(N / 4));
      const unsigned tot = E << lq;
      spmm_scatter<<<(tot + 255) / 256, 256, 0, stream>>>(
          P, esrc, edst, ew, agg, E, N, lq);
    }

    // Bias + activation.
    {
      const unsigned tot = (unsigned)NNODES * (unsigned)N;
      if (last)
        bias_f32<<<(tot + 255) / 256, 256, 0, stream>>>(agg, bl[l], tot, N);
      else
        bias_relu_bf16<<<(tot + 255) / 256, 256, 0, stream>>>(agg, bl[l], Hb,
                                                              tot, N);
    }
  }
}
